// MGACMDAAdapter_47210280518128
// MI455X (gfx1250) — matrix operand order repaired and checked by reading, not compiled, not run on hardware
//
#include <hip/hip_runtime.h>
#include <hip/hip_bf16.h>

typedef __attribute__((ext_vector_type(16))) __bf16 v16bf;
typedef __attribute__((ext_vector_type(8)))  __bf16 v8bf;
typedef __attribute__((ext_vector_type(8)))  float  v8f;

#define GAT_N     3072
#define GAT_INDIM 256
#define GAT_H1    128
#define GAT_H2    64
#define GAT_HEADS 4

// ---------------- WMMA fragment helpers (CDNA5 16x16x32 bf16 layouts) -------
// A (16x32, 16-bit): lane m = lane&15; half = lane>>4.
//   vector elem i -> K = (i<8) ? half*8+i : 16+half*8+(i-8)
// B (32x16, 16-bit, loaded from BT stored [O][K] row-major): elem i -> K same as A
// C/D (16x16 f32): vector elem r -> row = r + 8*half, col = lane&15

static __device__ __forceinline__ v16bf ld_a_frag(const __bf16* A, long lda,
                                                  int row, int kb, int half) {
  const __bf16* p = A + (long)row * lda + kb + half * 8;
  v8bf lo = *(const v8bf*)p;         // K = kb + half*8 .. +7
  v8bf hi = *(const v8bf*)(p + 16);  // K = kb + 16 + half*8 .. +7
  v16bf r;
#pragma unroll
  for (int i = 0; i < 8; ++i) { r[i] = lo[i]; r[8 + i] = hi[i]; }
  return r;
}

static __device__ __forceinline__ v16bf ld_b_frag(const __bf16* BT, long ldb,
                                                  int col, int kb, int half) {
  const __bf16* p = BT + (long)col * ldb + kb + half * 8;
  v8bf lo = *(const v8bf*)p;
  v8bf hi = *(const v8bf*)(p + 16);
  v16bf r;
#pragma unroll
  for (int i = 0; i < 8; ++i) { r[i] = lo[i]; r[8 + i] = hi[i]; }
  return r;
}

// Double-buffered GEMM inner loop: tiles for step kb+32 are loaded while the
// WMMAs for step kb execute; the tail issues a redundant (harmless) reload of
// tile 0 to keep the loop branch-free.
template <int OT>
static __device__ __forceinline__ void gemm_acc(const __bf16* A, long lda,
                                                const __bf16* BT, long ldb,
                                                int arow, int K, int n, int half,
                                                v8f* acc) {
  v16bf a = ld_a_frag(A, lda, arow, 0, half);
  v16bf b[OT];
#pragma unroll
  for (int t = 0; t < OT; ++t) b[t] = ld_b_frag(BT, ldb, t * 16 + n, 0, half);
  for (int kb = 0; kb < K; kb += 32) {
    int kn = (kb + 32 < K) ? kb + 32 : 0;
    v16bf an = ld_a_frag(A, lda, arow, kn, half);
    v16bf bn[OT];
#pragma unroll
    for (int t = 0; t < OT; ++t) bn[t] = ld_b_frag(BT, ldb, t * 16 + n, kn, half);
#pragma unroll
    for (int t = 0; t < OT; ++t)
      acc[t] = __builtin_amdgcn_wmma_f32_16x16x32_bf16(
          false, a, false, b[t], (short)0, acc[t], false, false);
    a = an;
#pragma unroll
    for (int t = 0; t < OT; ++t) b[t] = bn[t];
  }
}

// ---------------- format conversion ----------------------------------------
__global__ void k_f32_to_bf16(const float* __restrict__ in,
                              __bf16* __restrict__ out, long n) {
  long i = (long)blockIdx.x * blockDim.x + threadIdx.x;
  if (i < n) out[i] = (__bf16)in[i];
}

// in f32 [B][R][C] -> out bf16 [B][C][R]
__global__ void k_transpose_bf16(const float* __restrict__ in,
                                 __bf16* __restrict__ out, int B, int R, int C) {
  long i = (long)blockIdx.x * blockDim.x + threadIdx.x;
  long tot = (long)B * R * C;
  if (i >= tot) return;
  int c = (int)(i % C);
  long t = i / C;
  int r = (int)(t % R);
  int b = (int)(t / R);
  out[((long)b * C + c) * R + r] = (__bf16)in[i];
}

// ---------------- Wh = x @ W[h], stored transposed WhT[h][Fout][N] ----------
template <int OT>
__launch_bounds__(128, 1)
__global__ void k_gemm_wh(const __bf16* __restrict__ A, int K,
                          const __bf16* __restrict__ WT,  // [H][Fout][K]
                          __bf16* __restrict__ WhT,       // [H][Fout][N]
                          int N) {
  const int Fout = OT * 16;
  int lane = threadIdx.x & 31;
  int wid = threadIdx.x >> 5;
  int rowblk = blockIdx.x * (blockDim.x >> 5) + wid;
  int h = blockIdx.y;
  if (rowblk * 16 >= N) return;
  int m = lane & 15, half = lane >> 4;
  v8f acc[OT];
#pragma unroll
  for (int t = 0; t < OT; ++t)
#pragma unroll
    for (int r = 0; r < 8; ++r) acc[t][r] = 0.f;
  const __bf16* BT = WT + (long)h * Fout * K;
  gemm_acc<OT>(A, K, BT, K, rowblk * 16 + m, K, m, half, acc);
  __bf16* dstp = WhT + (long)h * Fout * N;
#pragma unroll
  for (int t = 0; t < OT; ++t)
#pragma unroll
    for (int r = 0; r < 8; ++r)
      dstp[(long)(t * 16 + m) * N + rowblk * 16 + r + 8 * half] =
          (__bf16)acc[t][r];
}

// ---------------- src/dst rank-1 score vectors ------------------------------
__global__ void k_srcdst(const __bf16* __restrict__ WhT,  // [H][Fout][N]
                         const float* __restrict__ as_, const float* __restrict__ ad_,
                         float* __restrict__ src, float* __restrict__ dst,
                         int N, int Fout, int H) {
  long i = (long)blockIdx.x * blockDim.x + threadIdx.x;
  if (i >= (long)H * N) return;
  int h = (int)(i / N);
  int n = (int)(i % N);
  float s = 0.f, d = 0.f;
  for (int o = 0; o < Fout; ++o) {
    float w = (float)WhT[((long)h * Fout + o) * N + n];
    s += w * as_[h * Fout + o];
    d += w * ad_[h * Fout + o];
  }
  src[i] = s;
  dst[i] = d;
}

// ---------------- fused masked softmax + P@Wh (flash style) -----------------
template <int OT>
__launch_bounds__(128, 1)
__global__ void k_attn(const __bf16* __restrict__ WhT,  // [H][Fout][N]
                       const float* __restrict__ src, const float* __restrict__ dstv,
                       const int* __restrict__ adj,
                       __bf16* __restrict__ hp,  // [N][HF] head-concat
                       int N, int HF) {
  const int Fout = OT * 16;
  int lane = threadIdx.x & 31;
  int wid = threadIdx.x >> 5;
  int rowblk = blockIdx.x * (blockDim.x >> 5) + wid;
  int h = blockIdx.y;
  if (rowblk * 16 >= N) return;
  int m = lane & 15, half = lane >> 4;
  int row = rowblk * 16 + m;

  float s_i = src[(long)h * N + row];
  const int* adjrow = adj + (long)row * N;
  const float* drow = dstv + (long)h * N;
  const __bf16* BT = WhT + (long)h * Fout * N;

  v8f acc[OT];
#pragma unroll
  for (int t = 0; t < OT; ++t)
#pragma unroll
    for (int r = 0; r < 8; ++r) acc[t][r] = 0.f;

  float runmax = -3.0e38f, runsum = 0.f;

  for (int jb = 0; jb < N; jb += 32) {
    // Issue the B-tile loads first: their latency overlaps the softmax VALU
    // work (leaky/mask/max/exp/shuffles) below.
    v16bf b[OT];
#pragma unroll
    for (int t = 0; t < OT; ++t) b[t] = ld_b_frag(BT, N, t * 16 + m, jb, half);

    float e[16];
    float tmax = -3.0e38f;
#pragma unroll
    for (int i = 0; i < 16; ++i) {
      int kk = (i < 8) ? (half * 8 + i) : (16 + half * 8 + (i - 8));
      int j = jb + kk;
      float v = s_i + drow[j];
      v = v > 0.f ? v : 0.2f * v;            // leaky_relu(0.2)
      v = (adjrow[j] > 0) ? v : -9.0e15f;    // adjacency mask
      e[i] = v;
      tmax = fmaxf(tmax, v);
    }
    // row reduction across the two half-lanes sharing this row
    tmax = fmaxf(tmax, __shfl_xor(tmax, 16, 32));
    float newmax = fmaxf(runmax, tmax);
    float scale = __expf(runmax - newmax);
    float lsum = 0.f;
    v16bf pa;
#pragma unroll
    for (int i = 0; i < 16; ++i) {
      float pv = __expf(e[i] - newmax);
      lsum += pv;
      pa[i] = (__bf16)pv;
    }
    lsum += __shfl_xor(lsum, 16, 32);
    runsum = runsum * scale + lsum;
    runmax = newmax;
    // rescale accumulators: row of acc elem r is r+8*half; lane x (<16) owns row x stats
    float sc[8];
#pragma unroll
    for (int r = 0; r < 8; ++r) sc[r] = __shfl(scale, r + 8 * half, 32);
#pragma unroll
    for (int t = 0; t < OT; ++t)
#pragma unroll
      for (int r = 0; r < 8; ++r) acc[t][r] *= sc[r];
#pragma unroll
    for (int t = 0; t < OT; ++t)
      acc[t] = __builtin_amdgcn_wmma_f32_16x16x32_bf16(
          false, pa, false, b[t], (short)0, acc[t], false, false);
  }

  float rs[8];
#pragma unroll
  for (int r = 0; r < 8; ++r) rs[r] = __shfl(runsum, r + 8 * half, 32);
#pragma unroll
  for (int t = 0; t < OT; ++t)
#pragma unroll
    for (int r = 0; r < 8; ++r) {
      float v = acc[t][r] / rs[r];
      v = v > 0.f ? v : (__expf(v) - 1.f);  // elu
      hp[(long)(rowblk * 16 + r + 8 * half) * HF + h * Fout + t * 16 + m] =
          (__bf16)v;
    }
}

// ---------------- out = relu(elu(hp@lin) + h@res) ---------------------------
template <int OT>
__launch_bounds__(128, 1)
__global__ void k_mix(const __bf16* __restrict__ hp, int HF,
                      const __bf16* __restrict__ linT,   // [Fout][HF]
                      const __bf16* __restrict__ hin, int Fin,
                      const __bf16* __restrict__ resT,   // [Fout][Fin]
                      __bf16* __restrict__ outb,         // may be null
                      float* __restrict__ outf,          // may be null
                      int N) {
  const int Fout = OT * 16;
  int lane = threadIdx.x & 31;
  int wid = threadIdx.x >> 5;
  int rowblk = blockIdx.x * (blockDim.x >> 5) + wid;
  if (rowblk * 16 >= N) return;
  int m = lane & 15, half = lane >> 4;
  int row = rowblk * 16 + m;

  v8f acc[OT];
#pragma unroll
  for (int t = 0; t < OT; ++t)
#pragma unroll
    for (int r = 0; r < 8; ++r) acc[t][r] = 0.f;

  gemm_acc<OT>(hp, HF, linT, HF, row, HF, m, half, acc);  // hp @ lin
#pragma unroll
  for (int t = 0; t < OT; ++t)
#pragma unroll
    for (int r = 0; r < 8; ++r) {
      float v = acc[t][r];
      acc[t][r] = v > 0.f ? v : (__expf(v) - 1.f);        // elu
    }
  gemm_acc<OT>(hin, Fin, resT, Fin, row, Fin, m, half, acc);  // + h @ res

#pragma unroll
  for (int t = 0; t < OT; ++t)
#pragma unroll
    for (int r = 0; r < 8; ++r) {
      float v = fmaxf(acc[t][r], 0.f);  // relu(relu(x)) == relu(x)
      long idx = (long)(rowblk * 16 + r + 8 * half) * Fout + t * 16 + m;
      if (outb) outb[idx] = (__bf16)v;
      if (outf) outf[idx] = v;
    }
}

// ---------------- launch ----------------------------------------------------
extern "C" void kernel_launch(void* const* d_in, const int* in_sizes, int n_in,
                              void* d_out, int out_size, void* d_ws, size_t ws_size,
                              hipStream_t stream) {
  (void)in_sizes; (void)n_in; (void)out_size; (void)ws_size;
  const float* x    = (const float*)d_in[0];
  const int*   adj  = (const int*)  d_in[1];
  const float* W1   = (const float*)d_in[2];
  const float* a1s  = (const float*)d_in[3];
  const float* a1d  = (const float*)d_in[4];
  const float* lin1 = (const float*)d_in[5];
  const float* res1 = (const float*)d_in[6];
  const float* W2   = (const float*)d_in[7];
  const float* a2s  = (const float*)d_in[8];
  const float* a2d  = (const float*)d_in[9];
  const float* lin2 = (const float*)d_in[10];
  const float* res2 = (const float*)d_in[11];

  char* p = (char*)d_ws;
  auto take = [&](size_t bytes) -> char* {
    char* r = p;
    p += (bytes + 255) & ~(size_t)255;
    return r;
  };
  __bf16* xb    = (__bf16*)take((size_t)GAT_N * GAT_INDIM * 2);
  __bf16* w1t   = (__bf16*)take((size_t)GAT_HEADS * GAT_H1 * GAT_INDIM * 2);
  __bf16* wh1t  = (__bf16*)take((size_t)GAT_HEADS * GAT_H1 * GAT_N * 2);
  float*  src1  = (float*) take((size_t)GAT_HEADS * GAT_N * 4);
  float*  dst1  = (float*) take((size_t)GAT_HEADS * GAT_N * 4);
  __bf16* hp1   = (__bf16*)take((size_t)GAT_N * GAT_HEADS * GAT_H1 * 2);
  __bf16* lin1t = (__bf16*)take((size_t)GAT_H1 * GAT_HEADS * GAT_H1 * 2);
  __bf16* res1t = (__bf16*)take((size_t)GAT_H1 * GAT_INDIM * 2);
  __bf16* h1b   = (__bf16*)take((size_t)GAT_N * GAT_H1 * 2);
  __bf16* w2t   = (__bf16*)take((size_t)GAT_HEADS * GAT_H2 * GAT_H1 * 2);
  __bf16* wh2t  = (__bf16*)take((size_t)GAT_HEADS * GAT_H2 * GAT_N * 2);
  float*  src2  = (float*) take((size_t)GAT_HEADS * GAT_N * 4);
  float*  dst2  = (float*) take((size_t)GAT_HEADS * GAT_N * 4);
  __bf16* hp2   = (__bf16*)take((size_t)GAT_N * GAT_HEADS * GAT_H2 * 2);
  __bf16* lin2t = (__bf16*)take((size_t)GAT_H2 * GAT_HEADS * GAT_H2 * 2);
  __bf16* res2t = (__bf16*)take((size_t)GAT_H2 * GAT_H1 * 2);

  // format conversion / pre-transpose of all B operands
  k_f32_to_bf16<<<(GAT_N * GAT_INDIM + 255) / 256, 256, 0, stream>>>(
      x, xb, (long)GAT_N * GAT_INDIM);
  k_transpose_bf16<<<(GAT_HEADS * GAT_INDIM * GAT_H1 + 255) / 256, 256, 0, stream>>>(
      W1, w1t, GAT_HEADS, GAT_INDIM, GAT_H1);
  k_transpose_bf16<<<(GAT_HEADS * GAT_H1 * GAT_H1 + 255) / 256, 256, 0, stream>>>(
      lin1, lin1t, 1, GAT_HEADS * GAT_H1, GAT_H1);
  k_transpose_bf16<<<(GAT_INDIM * GAT_H1 + 255) / 256, 256, 0, stream>>>(
      res1, res1t, 1, GAT_INDIM, GAT_H1);
  k_transpose_bf16<<<(GAT_HEADS * GAT_H1 * GAT_H2 + 255) / 256, 256, 0, stream>>>(
      W2, w2t, GAT_HEADS, GAT_H1, GAT_H2);
  k_transpose_bf16<<<(GAT_HEADS * GAT_H2 * GAT_H2 + 255) / 256, 256, 0, stream>>>(
      lin2, lin2t, 1, GAT_HEADS * GAT_H2, GAT_H2);
  k_transpose_bf16<<<(GAT_H1 * GAT_H2 + 255) / 256, 256, 0, stream>>>(
      res2, res2t, 1, GAT_H1, GAT_H2);

  dim3 blk(128);                       // 4 waves, one 16-row tile each
  dim3 gh(GAT_N / 16 / 4, GAT_HEADS);  // (48, 4)
  int gm = GAT_N / 16 / 4;             // 48

  // ---- layer 1 ----
  k_gemm_wh<8><<<gh, blk, 0, stream>>>(xb, GAT_INDIM, w1t, wh1t, GAT_N);
  k_srcdst<<<(GAT_HEADS * GAT_N + 255) / 256, 256, 0, stream>>>(
      wh1t, a1s, a1d, src1, dst1, GAT_N, GAT_H1, GAT_HEADS);
  k_attn<8><<<gh, blk, 0, stream>>>(wh1t, src1, dst1, adj, hp1, GAT_N,
                                    GAT_HEADS * GAT_H1);
  k_mix<8><<<gm, blk, 0, stream>>>(hp1, GAT_HEADS * GAT_H1, lin1t, xb,
                                   GAT_INDIM, res1t, h1b, (float*)nullptr,
                                   GAT_N);

  // ---- layer 2 ----
  k_gemm_wh<4><<<gh, blk, 0, stream>>>(h1b, GAT_H1, w2t, wh2t, GAT_N);
  k_srcdst<<<(GAT_HEADS * GAT_N + 255) / 256, 256, 0, stream>>>(
      wh2t, a2s, a2d, src2, dst2, GAT_N, GAT_H2, GAT_HEADS);
  k_attn<4><<<gh, blk, 0, stream>>>(wh2t, src2, dst2, adj, hp2, GAT_N,
                                    GAT_HEADS * GAT_H2);
  k_mix<4><<<gm, blk, 0, stream>>>(hp2, GAT_HEADS * GAT_H2, lin2t, h1b,
                                   GAT_H1, res2t, (__bf16*)nullptr,
                                   (float*)d_out, GAT_N);
}
